// Attention_87308095193108
// MI455X (gfx1250) — matrix operand order repaired and checked by reading, not compiled, not run on hardware
//
#include <hip/hip_runtime.h>

typedef __attribute__((ext_vector_type(16))) _Float16 v16h;
typedef __attribute__((ext_vector_type(8)))  float    v8f;

#define WAVES   8
#define QTILE   16     // rows per WMMA tile
#define NQT     2      // q-tiles per wave -> 32 rows per wave
#define KCHUNK  64
#define DH      64
#define KS      72     // padded row stride (halves) for K chunk in LDS
#define VS      72     // padded row stride for V^T in LDS
#define PS      72     // padded row stride for P scratch in LDS

union AFrag { v16h v; _Float16 h[16]; };

// max with lane (lane ^ MASK) within each 16-lane DPP row: pure VALU, no LDS
template <int MASK>
__device__ __forceinline__ float fmax_xor(float v) {
  int p = __builtin_amdgcn_update_dpp(__float_as_int(v), __float_as_int(v),
                                      0x160 | MASK /* ROW_XMASK */, 0xf, 0xf, true);
  return fmaxf(v, __int_as_float(p));
}
__device__ __forceinline__ float rowmax16(float v) {
  v = fmax_xor<1>(v);
  v = fmax_xor<2>(v);
  v = fmax_xor<4>(v);
  v = fmax_xor<8>(v);
  return v;
}

__global__ __launch_bounds__(256)
void fattn_wmma(const float* __restrict__ Q, const float* __restrict__ K,
                const float* __restrict__ V, float* __restrict__ O,
                int S) {
  __shared__ _Float16 Kh[KCHUNK * KS];        // K chunk, f16, row-major [key][d]
  __shared__ _Float16 Vt[DH * VS];            // V chunk, f16, transposed [d][key]
  __shared__ _Float16 Pl[WAVES * QTILE * PS]; // per-wave P scratch [row][key]

  const int tid  = threadIdx.x;
  const int wave = tid >> 5;
  const int lane = tid & 31;
  const int m    = lane & 15;   // row (A) or column (B/C) index within 16
  const int hi   = lane >> 4;   // lane-half selector

  const int rows_per_blk = WAVES * QTILE * NQT;      // 256
  const int qblocks = S / rows_per_blk;              // 16
  const int b  = blockIdx.x / qblocks;
  const int qb = blockIdx.x % qblocks;
  const int q0 = qb * rows_per_blk + wave * (QTILE * NQT);

  const float* Qb = Q + ((size_t)b * S + q0) * DH;
  const float* Kb = K + (size_t)b * S * DH;
  const float* Vb = V + (size_t)b * S * DH;
  float*       Ob = O + ((size_t)b * S + q0) * DH;

  const float scale = 0.125f; // 1/sqrt(64)

  // ---- Q tiles -> 16x32 f16 A-fragments (pre-scaled) ----
  // A layout: lane m holds row m; h[i]: K = c*32 + hi*16 + i
  AFrag aq[NQT][2];
#pragma unroll
  for (int t = 0; t < NQT; ++t)
#pragma unroll
    for (int c = 0; c < 2; ++c) {
      const float* qrow = Qb + (size_t)(t * QTILE + m) * DH + c * 32 + hi * 16;
#pragma unroll
      for (int i = 0; i < 8; ++i) {
        aq[t][c].h[i]     = (_Float16)(qrow[i]      * scale);
        aq[t][c].h[i + 8] = (_Float16)(qrow[8 + i] * scale);
      }
    }

  // all-ones B fragment: rowsum(P) = P x ones via WMMA
  AFrag ones;
#pragma unroll
  for (int i = 0; i < 16; ++i) ones.h[i] = (_Float16)1.0f;

  // ---- flash state: [tile][register r] <-> query row (t*16 + r + hi*8) ----
  v8f o[NQT][4] = {};
  float rm[NQT][8], rl[NQT][8];
#pragma unroll
  for (int t = 0; t < NQT; ++t)
#pragma unroll
    for (int r = 0; r < 8; ++r) { rm[t][r] = -__builtin_huge_valf(); rl[t][r] = 0.0f; }

  const int nchunks = S / KCHUNK;
  for (int ch = 0; ch < nchunks; ++ch) {
    const int ko = ch * KCHUNK;

    // ---- stage K (row-major) and V (transposed) chunks as f16 ----
    {
      const int row = tid >> 2;          // 0..63 key row
      const int col = (tid & 3) * 16;    // 0,16,32,48 d offset
      const float* kp = Kb + (size_t)(ko + row) * DH + col;
      const float* vp = Vb + (size_t)(ko + row) * DH + col;
#pragma unroll
      for (int j = 0; j < 16; ++j) {
        Kh[row * KS + col + j]   = (_Float16)kp[j];
        Vt[(col + j) * VS + row] = (_Float16)vp[j];
      }
    }
    if (ch + 1 < nchunks) {
      const char* nk = (const char*)(Kb + (size_t)(ko + KCHUNK) * DH);
      const char* nv = (const char*)(Vb + (size_t)(ko + KCHUNK) * DH);
      __builtin_prefetch(nk + tid * 64, 0, 3);
      __builtin_prefetch(nv + tid * 64, 0, 3);
    }
    __syncthreads();

    // ---- per q-tile pipeline (keeps only one tile's scores live) ----
#pragma unroll
    for (int t = 0; t < NQT; ++t) {
      // S = Q K^T : four 16x16 column tiles, K-dim(d) 64 = 2 x wmma(k=32)
      v8f s[4];
#pragma unroll
      for (int nt = 0; nt < 4; ++nt) {
        // B layout: lane column n = m; h[i]: K(d) = kc*32 + hi*16 + i
        AFrag bk0, bk1;
        const _Float16* krow = &Kh[(nt * 16 + m) * KS + hi * 16];
#pragma unroll
        for (int i = 0; i < 16; ++i) { bk0.h[i] = krow[i]; bk1.h[i] = krow[32 + i]; }
        v8f z = {};
        v8f tt = __builtin_amdgcn_wmma_f32_16x16x32_f16(false, aq[t][0].v, false, bk0.v,
                                                        (short)0, z, false, false);
        s[nt]  = __builtin_amdgcn_wmma_f32_16x16x32_f16(false, aq[t][1].v, false, bk1.v,
                                                        (short)0, tt, false, false);
      }

      // new row maxima via DPP row_xmask butterfly (VALU only)
      float nmv[8];
#pragma unroll
      for (int r = 0; r < 8; ++r) {
        float cmax = rowmax16(fmaxf(fmaxf(s[0][r], s[1][r]), fmaxf(s[2][r], s[3][r])));
        nmv[r] = fmaxf(rm[t][r], cmax);
      }

      // wave-uniform vote: rescale state only when some row's max grew.
      // Scalar branch (EXEC stays all-1s, WMMA-safe); ~1/chunk probability late in the scan.
      bool upd = false;
#pragma unroll
      for (int r = 0; r < 8; ++r) upd = upd || (nmv[r] > rm[t][r]);
      if (__ballot(upd)) {
#pragma unroll
        for (int r = 0; r < 8; ++r) {
          const float alpha = __expf(rm[t][r] - nmv[r]);
          rm[t][r] = nmv[r];
          rl[t][r] *= alpha;
#pragma unroll
          for (int dt = 0; dt < 4; ++dt) o[t][dt][r] *= alpha;
        }
      }

      // P = exp(s - nm) -> LDS in C-layout: row (r + hi*8), col (nt*16 + m)
#pragma unroll
      for (int r = 0; r < 8; ++r) {
        _Float16* prow = &Pl[(wave * QTILE + r + hi * 8) * PS + m];
#pragma unroll
        for (int nt = 0; nt < 4; ++nt)
          prow[nt * 16] = (_Float16)__expf(s[nt][r] - nmv[r]);
      }
      // intra-wave cross-lane LDS write -> read ordering
      asm volatile("s_wait_dscnt 0" ::: "memory");

      // reload P in A-layout: lane row m; h[i]: key = kc*32 + hi*16 + i
      AFrag ap[2];
#pragma unroll
      for (int kc = 0; kc < 2; ++kc) {
        const _Float16* prow = &Pl[(wave * QTILE + m) * PS + kc * 32 + hi * 16];
#pragma unroll
        for (int i = 0; i < 8; ++i) { ap[kc].h[i] = prow[i]; ap[kc].h[i + 8] = prow[8 + i]; }
      }

      // row sums of P via WMMA with all-ones B
      {
        v8f z = {};
        v8f t0 = __builtin_amdgcn_wmma_f32_16x16x32_f16(false, ap[0].v, false, ones.v,
                                                        (short)0, z, false, false);
        v8f rs = __builtin_amdgcn_wmma_f32_16x16x32_f16(false, ap[1].v, false, ones.v,
                                                        (short)0, t0, false, false);
#pragma unroll
        for (int r = 0; r < 8; ++r) rl[t][r] += rs[r];
      }

      // O += P V : 4 column tiles of 16 over D=64, K-dim 64 keys
#pragma unroll
      for (int dt = 0; dt < 4; ++dt) {
#pragma unroll
        for (int kc = 0; kc < 2; ++kc) {
          // B layout: lane column (d) = m; h[i]: key = kc*32 + hi*16 + i
          AFrag bv;
          const _Float16* vtr = &Vt[(dt * 16 + m) * VS + kc * 32 + hi * 16];
#pragma unroll
          for (int i = 0; i < 16; ++i) bv.h[i] = vtr[i];
          o[t][dt] = __builtin_amdgcn_wmma_f32_16x16x32_f16(false, ap[kc].v, false, bv.v,
                                                            (short)0, o[t][dt], false, false);
        }
      }
    }
    __syncthreads();
  }

  // ---- epilogue: O / l, store ----
#pragma unroll
  for (int t = 0; t < NQT; ++t)
#pragma unroll
    for (int r = 0; r < 8; ++r) {
      const float inv = 1.0f / rl[t][r];
      const int row = t * QTILE + r + hi * 8;
#pragma unroll
      for (int dt = 0; dt < 4; ++dt)
        Ob[(size_t)row * DH + dt * 16 + m] = o[t][dt][r] * inv;
    }
}

extern "C" void kernel_launch(void* const* d_in, const int* in_sizes, int n_in,
                              void* d_out, int out_size, void* d_ws, size_t ws_size,
                              hipStream_t stream) {
  (void)in_sizes; (void)n_in; (void)d_ws; (void)ws_size; (void)out_size;
  const float* Q = (const float*)d_in[0];
  const float* K = (const float*)d_in[1];
  const float* V = (const float*)d_in[2];
  float* O = (float*)d_out;

  const int B = 16, S = 4096;
  dim3 grid(B * S / (WAVES * QTILE * NQT));  // 256 blocks, 32 q-rows per wave
  dim3 block(256);                            // 8 waves (wave32)
  fattn_wmma<<<grid, block, 0, stream>>>(Q, K, V, O, S);
}
